// MultiheadAttention_9105330667672
// MI455X (gfx1250) — matrix operand order repaired and checked by reading, not compiled, not run on hardware
//
#include <hip/hip_runtime.h>
#include <hip/hip_bf16.h>

// ---------------- CDNA5 WMMA plumbing ----------------
typedef __attribute__((ext_vector_type(16))) __bf16 v16bf;
typedef __attribute__((ext_vector_type(8)))  float  v8f;

union ABFrag {           // one A or B operand: 16 bf16 per lane (8 VGPRs)
  v16bf v;
  uint4 q[2];
};
union CFrag {            // 16x16 f32 accumulator: 8 VGPRs per lane
  v8f   v;
  float f[8];
};

__device__ __forceinline__ unsigned short f2bf(float x) {
  unsigned int u = __float_as_uint(x);
  u += 0x7fffu + ((u >> 16) & 1u);        // round-to-nearest-even
  return (unsigned short)(u >> 16);
}

// A matrix 16x32 bf16, memory row-major (M,K), lda elements/row.
// Lane l: row = l%16; K chunks [kb, kb+8) and [kb+16, kb+24), kb = (l<16?0:8).
__device__ __forceinline__ ABFrag load_a_frag(const unsigned short* base, int lda) {
  int lane = threadIdx.x & 31;
  int r  = lane & 15;
  int kb = (lane < 16) ? 0 : 8;
  const unsigned short* p = base + r * lda + kb;
  ABFrag f;
  f.q[0] = *(const uint4*)(p);
  f.q[1] = *(const uint4*)(p + 16);
  return f;
}

// B matrix 32x16 bf16 (KxN); memory holds (N,K) row-major (i.e. B^T), ldb elems/row.
// Lane l: col = l%16; K chunks [kb, kb+8) and [kb+16, kb+24), kb = (l<16?0:8).
__device__ __forceinline__ ABFrag load_b_frag(const unsigned short* base, int ldb) {
  int lane = threadIdx.x & 31;
  int c  = lane & 15;
  int kb = (lane < 16) ? 0 : 8;
  const unsigned short* p = base + c * ldb + kb;
  ABFrag f;
  f.q[0] = *(const uint4*)(p);
  f.q[1] = *(const uint4*)(p + 16);
  return f;
}

__device__ __forceinline__ v8f wmma_bf16(ABFrag a, ABFrag b, v8f c) {
  return __builtin_amdgcn_wmma_f32_16x16x32_bf16(false, a.v, false, b.v,
                                                 (short)0, c, false, false);
}

// ---------------- problem constants ----------------
constexpr int NB   = 512;   // batch
constexpr int NS   = 256;   // states (channels)
constexpr int NP   = 196;   // pixels
constexpr int PPAD = 208;   // pixels padded to 13*16
constexpr int NH   = 4;     // heads
constexpr int ND   = 64;    // head dim

// ---------------- kernel 1: weights f32 -> bf16 (wq|wk|wv packed 768x256) ----
__global__ void wconv_kernel(const float* __restrict__ wq, const float* __restrict__ wk,
                             const float* __restrict__ wv, const float* __restrict__ wo,
                             unsigned short* __restrict__ wqkv, unsigned short* __restrict__ wobf) {
  int idx = blockIdx.x * blockDim.x + threadIdx.x;   // 0..65535
  wqkv[idx]          = f2bf(wq[idx]);
  wqkv[idx + 65536]  = f2bf(wk[idx]);
  wqkv[idx + 131072] = f2bf(wv[idx]);
  wobf[idx]          = f2bf(wo[idx]);
}

// ---------------- kernel 2: x (B,S,P) f32 -> xT (B,PPAD,S) bf16, zero pad ----
__global__ __launch_bounds__(256) void xpose_kernel(const float* __restrict__ x,
                                                    unsigned short* __restrict__ xT) {
  __shared__ float tile[NS][17];                    // [channel][pixel], padded pitch
  int b  = blockIdx.y;
  int p0 = blockIdx.x * 16;
  int t  = threadIdx.x;
  int pc = t & 15;
  int c0 = t >> 4;
#pragma unroll
  for (int j = 0; j < 16; ++j) {
    int c = c0 * 16 + j;
    int p = p0 + pc;
    float val = 0.f;
    if (p < NP) val = x[((size_t)b * NS + c) * NP + p];
    tile[c][pc] = val;
  }
  __syncthreads();
  int pr = t >> 4;                                  // row of xT within tile
  int cc = (t & 15) * 16;
  size_t dst = ((size_t)b * PPAD + (p0 + pr)) * NS + cc;
#pragma unroll
  for (int i = 0; i < 16; ++i) xT[dst + i] = f2bf(tile[cc + i][pr]);
}

// ---------------- kernel 3: fused QKV projection --------------------------
// y[o][p] = sum_c wqkv[o][c] * xT[b][p][c] + bias ; o in [0,768)
// q,k stored (B,H,PPAD,D);  v stored transposed (B,H,D,PPAD)
__global__ __launch_bounds__(256) void qkv_kernel(const unsigned short* __restrict__ xT,
                                                  const unsigned short* __restrict__ wqkv,
                                                  const float* __restrict__ bq,
                                                  const float* __restrict__ bk,
                                                  const float* __restrict__ bv,
                                                  unsigned short* __restrict__ qw,
                                                  unsigned short* __restrict__ kw,
                                                  unsigned short* __restrict__ vtw) {
  __shared__ unsigned short bpan[16 * NS];          // 16 pixel-rows, full K (8 KB)
  int b   = blockIdx.y;
  int tid = blockIdx.x;                             // 0..77
  int ot  = tid / 13, pt = tid % 13;
  int o0  = ot * 128, p0 = pt * 16;
  int t   = threadIdx.x;

  const unsigned int* src = (const unsigned int*)(xT + ((size_t)b * PPAD + p0) * NS);
  unsigned int* dstp = (unsigned int*)bpan;
#pragma unroll
  for (int j = 0; j < 8; ++j) dstp[t + 256 * j] = src[t + 256 * j];
  __syncthreads();

  int wave = t >> 5;
  int orow = o0 + wave * 16;                        // 16 output rows per wave
  CFrag acc;
#pragma unroll
  for (int i = 0; i < 8; ++i) acc.f[i] = 0.f;
#pragma unroll
  for (int ks = 0; ks < 8; ++ks) {
    ABFrag a  = load_a_frag(wqkv + (size_t)orow * NS + ks * 32, NS);
    ABFrag bf = load_b_frag(bpan + ks * 32, NS);
    acc.v = wmma_bf16(a, bf, acc.v);
  }

  int lane = t & 31;
  int hl   = lane >> 4;
  int col  = lane & 15;
  int p    = p0 + col;
  int proj  = orow >> 8;                            // 0=q 1=k 2=v, wave-uniform
  int obase = orow & 255;
  int h     = obase >> 6;                           // wave-uniform
  const float* bias = (proj == 0) ? bq : (proj == 1) ? bk : bv;
#pragma unroll
  for (int i = 0; i < 8; ++i) {
    int oo = obase + i + hl * 8;
    float y = acc.f[i] + bias[oo];
    int d = oo & 63;
    unsigned short v16 = f2bf(y);
    if (proj == 2) {
      vtw[(((size_t)b * NH + h) * ND + d) * PPAD + p] = v16;
    } else {
      unsigned short* dst = (proj == 0) ? qw : kw;
      dst[(((size_t)b * NH + h) * PPAD + p) * ND + d] = v16;
    }
  }
}

// ---------------- kernel 4: attention per (b,h) ---------------------------
__global__ __launch_bounds__(256) void attn_kernel(const unsigned short* __restrict__ qw,
                                                   const unsigned short* __restrict__ kw,
                                                   const unsigned short* __restrict__ vtw,
                                                   unsigned short* __restrict__ avbuf) {
  __shared__ unsigned short attlds[8][16 * 224];    // per-wave att strip, 56 KB
  int bh = blockIdx.x;
  int b = bh >> 2, h = bh & 3;
  int t = threadIdx.x;
  int wave = t >> 5, lane = t & 31;
  int hl = lane >> 4, col = lane & 15;

  const unsigned short* kbase  = kw  + (size_t)bh * PPAD * ND;
  const unsigned short* vtbase = vtw + (size_t)bh * ND * PPAD;

  for (int rep = 0; rep < 2; ++rep) {
    int it = wave + rep * 8;                        // i-tile index, wave-uniform
    if (it >= 13) break;
    int i0 = it * 16;
    const unsigned short* qbase = qw + ((size_t)bh * PPAD + i0) * ND;
    ABFrag a0 = load_a_frag(qbase, ND);             // K = 0..31 of d
    ABFrag a1 = load_a_frag(qbase + 32, ND);        // K = 32..63

    CFrag sc[13];
#pragma unroll
    for (int jt = 0; jt < 13; ++jt) {
#pragma unroll
      for (int i = 0; i < 8; ++i) sc[jt].f[i] = 0.f;
      ABFrag b0 = load_b_frag(kbase + (jt * 16) * ND, ND);
      ABFrag b1 = load_b_frag(kbase + (jt * 16) * ND + 32, ND);
      sc[jt].v = wmma_bf16(a0, b0, sc[jt].v);
      sc[jt].v = wmma_bf16(a1, b1, sc[jt].v);
    }

    // scale 1/sqrt(64) and mask padded key columns
#pragma unroll
    for (int jt = 0; jt < 13; ++jt) {
      int j = jt * 16 + col;
#pragma unroll
      for (int i = 0; i < 8; ++i) {
        float v = sc[jt].f[i] * 0.125f;
        sc[jt].f[i] = (j < NP) ? v : -3.0e38f;
      }
    }

    // row softmax: each VGPR slot i is one row spread over 16 lanes x 13 tiles
#pragma unroll
    for (int i = 0; i < 8; ++i) {
      float m = -3.0e38f;
#pragma unroll
      for (int jt = 0; jt < 13; ++jt) m = fmaxf(m, sc[jt].f[i]);
#pragma unroll
      for (int ofs = 1; ofs < 16; ofs <<= 1) m = fmaxf(m, __shfl_xor(m, ofs, 32));
      float s = 0.f;
#pragma unroll
      for (int jt = 0; jt < 13; ++jt) {
        float e = __expf(sc[jt].f[i] - m);
        sc[jt].f[i] = e;
        s += e;
      }
#pragma unroll
      for (int ofs = 1; ofs < 16; ofs <<= 1) s += __shfl_xor(s, ofs, 32);
      float inv = 1.0f / s;
#pragma unroll
      for (int jt = 0; jt < 13; ++jt) sc[jt].f[i] *= inv;
    }

    // re-fragment: write att strip (16 x 224 bf16) to per-wave LDS
    unsigned short* aw = attlds[wave];
#pragma unroll
    for (int jt = 0; jt < 13; ++jt) {
      int j = jt * 16 + col;
#pragma unroll
      for (int i = 0; i < 8; ++i) {
        int r = i + hl * 8;
        aw[r * 224 + j] = f2bf(sc[jt].f[i]);
      }
    }
    for (int idx = lane; idx < 256; idx += 32) {    // zero K padding j=208..223
      int r = idx >> 4, c = 208 + (idx & 15);
      aw[r * 224 + c] = 0;
    }

    // av[i][d] = sum_j att[i][j] * vT[d][j]   (16 x 64)
    CFrag av[4];
#pragma unroll
    for (int n = 0; n < 4; ++n)
#pragma unroll
      for (int i = 0; i < 8; ++i) av[n].f[i] = 0.f;
#pragma unroll
    for (int ks = 0; ks < 7; ++ks) {                // K = 224 (att zero past 207)
      ABFrag af = load_a_frag(aw + ks * 32, 224);
#pragma unroll
      for (int n = 0; n < 4; ++n) {
        ABFrag bf = load_b_frag(vtbase + (n * 16) * PPAD + ks * 32, PPAD);
        av[n].v = wmma_bf16(af, bf, av[n].v);
      }
    }

    // store with reference's no-head-transpose merge: flat = h*P*D + p*D + d,
    // merged buffer is (B, PPAD, S) with [pp][ss] = flat/256, flat%256
#pragma unroll
    for (int n = 0; n < 4; ++n) {
      int d = n * 16 + col;
#pragma unroll
      for (int i = 0; i < 8; ++i) {
        int p = i0 + i + hl * 8;
        if (p < NP) {
          int flat = h * (NP * ND) + p * ND + d;
          int pp = flat >> 8, ss = flat & 255;
          avbuf[((size_t)b * PPAD + pp) * NS + ss] = f2bf(av[n].f[i]);
        }
      }
    }
  }
}

// ---------------- kernel 5: output projection -----------------------------
// out[b][o][p] = sum_s wo[o][s] * merged[b][p][s] + bo[o]
__global__ __launch_bounds__(256) void oproj_kernel(const unsigned short* __restrict__ avbuf,
                                                    const unsigned short* __restrict__ wobf,
                                                    const float* __restrict__ bo,
                                                    float* __restrict__ out) {
  __shared__ unsigned short bpan[16 * NS];
  int b   = blockIdx.y;
  int tid = blockIdx.x;                             // 0..25
  int ot  = tid / 13, pt = tid % 13;
  int o0  = ot * 128, p0 = pt * 16;
  int t   = threadIdx.x;

  const unsigned int* src = (const unsigned int*)(avbuf + ((size_t)b * PPAD + p0) * NS);
  unsigned int* dstp = (unsigned int*)bpan;
#pragma unroll
  for (int j = 0; j < 8; ++j) dstp[t + 256 * j] = src[t + 256 * j];
  __syncthreads();

  int wave = t >> 5;
  int orow = o0 + wave * 16;
  CFrag acc;
#pragma unroll
  for (int i = 0; i < 8; ++i) acc.f[i] = 0.f;
#pragma unroll
  for (int ks = 0; ks < 8; ++ks) {
    ABFrag a  = load_a_frag(wobf + (size_t)orow * NS + ks * 32, NS);
    ABFrag bf = load_b_frag(bpan + ks * 32, NS);
    acc.v = wmma_bf16(a, bf, acc.v);
  }

  int lane = t & 31, hl = lane >> 4, col = lane & 15;
  int p = p0 + col;
  if (p < NP) {
#pragma unroll
    for (int i = 0; i < 8; ++i) {
      int o = orow + i + hl * 8;
      out[((size_t)b * NS + o) * NP + p] = acc.f[i] + bo[o];
    }
  }
}

// ---------------- launcher -------------------------------------------------
extern "C" void kernel_launch(void* const* d_in, const int* in_sizes, int n_in,
                              void* d_out, int out_size, void* d_ws, size_t ws_size,
                              hipStream_t stream) {
  (void)in_sizes; (void)n_in; (void)out_size; (void)ws_size;
  const float* x  = (const float*)d_in[0];
  const float* wq = (const float*)d_in[1];
  const float* bq = (const float*)d_in[2];
  const float* wk = (const float*)d_in[3];
  const float* bk = (const float*)d_in[4];
  const float* wv = (const float*)d_in[5];
  const float* bv = (const float*)d_in[6];
  const float* wo = (const float*)d_in[7];
  const float* bo = (const float*)d_in[8];
  float* out = (float*)d_out;

  // workspace layout (bf16 ushorts); xT region is reused as the merged (av) buffer
  const size_t SZ = (size_t)NB * PPAD * NS;         // 27,262,976 elems = 54.5 MB each
  unsigned short* xT   = (unsigned short*)d_ws;     // stage A; later: merged av
  unsigned short* qw   = xT + SZ;
  unsigned short* kw   = qw + SZ;
  unsigned short* vtw  = kw + SZ;
  unsigned short* wqkv = vtw + SZ;                  // 768x256
  unsigned short* wobf = wqkv + 768 * 256;          // 256x256

  wconv_kernel<<<256, 256, 0, stream>>>(wq, wk, wv, wo, wqkv, wobf);
  xpose_kernel<<<dim3(13, NB), 256, 0, stream>>>(x, xT);
  qkv_kernel<<<dim3(78, NB), 256, 0, stream>>>(xT, wqkv, bq, bk, bv, qw, kw, vtw);
  attn_kernel<<<NB * NH, 256, 0, stream>>>(qw, kw, vtw, xT /* reuse as merged */);
  oproj_kernel<<<dim3(26, NB), 256, 0, stream>>>(xT, wobf, bo, out);
}
